// Users_sets_Aggregator_51092930953376
// MI455X (gfx1250) — matrix operand order repaired and checked by reading, not compiled, not run on hardware
//
#include <hip/hip_runtime.h>
#include <hip/hip_bf16.h>

typedef _Float16 __attribute__((ext_vector_type(16))) v16h;
typedef _Float16 __attribute__((ext_vector_type(8)))  v8h;
typedef float    __attribute__((ext_vector_type(8)))  v8f;

#define BATCH 4096
#define LHIST 64
#define GMEM  8
#define DDIM  128

// ---------------------------------------------------------------------------
// ws layout (units of _Float16):
//   [0      , 32768 )  W1t  [128][256]
//   [32768  , 49152 )  W2t  [128][128]
//   [49152  , 81920 )  A1t  [128][256]
//   [81920  , 98304 )  A2t  [128][128]
//   [98304  , 114688)  GEt  [128][128]
//   [114688 , 638976)  Gall [4096][128]   (group embeddings, f16)
// ---------------------------------------------------------------------------

__global__ __launch_bounds__(256)
void transpose_f32_to_f16(const float* __restrict__ W, _Float16* __restrict__ Wt,
                          int K, int N) {
    int i = blockIdx.x * 256 + threadIdx.x;     // over K*N, row-major source
    if (i < K * N) {
        int k = i / N, n = i % N;
        Wt[(size_t)n * K + k] = (_Float16)W[i];
    }
}

// Load A fragment (16x32 f16): lane holds row m=(lane&15); K chunks {+0..7,+16..23},
// upper lanes offset by +8.  Source is K-contiguous row-major.
__device__ __forceinline__ v16h load_fragA(const _Float16* row, int k0, int lane) {
    int c0 = k0 + ((lane & 16) >> 1);
    v16h a;
    ((v8h*)&a)[0] = *(const v8h*)(row + c0);
    ((v8h*)&a)[1] = *(const v8h*)(row + c0 + 16);
    return a;
}

// Load B fragment (32x16 f16): lane holds col n=(lane&15); K chunks {+0..7,+16..23},
// upper lanes offset by +8.  Source is K-contiguous column-major (W transposed).
__device__ __forceinline__ v16h load_fragB(const _Float16* col, int k0, int lane) {
    int c0 = k0 + ((lane & 16) >> 1);
    v16h b;
    ((v8h*)&b)[0] = *(const v8h*)(col + c0);
    ((v8h*)&b)[1] = *(const v8h*)(col + c0 + 16);
    return b;
}

// One 64xK * Kx128 GEMM tile job per workgroup: wave w owns N-tile w (16 cols)
// across all 4 M-tiles; B fragments loaded once, reused over 4 accumulators.
// All 4 A fragments are loaded before the WMMA chain so the in-order DS
// returns allow staged s_wait_dscnt instead of a full drain per WMMA.
// Output: relu(acc + bias) stored as f16 row-major with stride out_stride.
__device__ __forceinline__ void gemm_relu_store(
        const _Float16* Xin, int in_stride, int K,
        const _Float16* __restrict__ Wt, const float* __restrict__ bias,
        _Float16* Xout, int out_stride, int wave, int lane) {
    int ncol = wave * 16 + (lane & 15);
    const _Float16* wp = Wt + (size_t)ncol * K;
    v8f acc[4] = {};
    for (int k0 = 0; k0 < K; k0 += 32) {
        v16h bf = load_fragB(wp, k0, lane);
        v16h af[4];
#pragma unroll
        for (int mt = 0; mt < 4; ++mt)
            af[mt] = load_fragA(Xin + (mt * 16 + (lane & 15)) * in_stride, k0, lane);
#pragma unroll
        for (int mt = 0; mt < 4; ++mt)
            acc[mt] = __builtin_amdgcn_wmma_f32_16x16x32_f16(
                false, af[mt], false, bf, (short)0, acc[mt], false, false);
    }
    float bv = bias[ncol];
#pragma unroll
    for (int mt = 0; mt < 4; ++mt) {
        int mb = mt * 16 + ((lane & 16) >> 1);   // C/D layout: upper lanes -> M+8
#pragma unroll
        for (int r = 0; r < 8; ++r)
            Xout[(size_t)(mb + r) * out_stride + ncol] =
                (_Float16)fmaxf(acc[mt][r] + bv, 0.0f);
    }
}

// ---------------------------------------------------------------------------
// Kernel 1: group representation  g = relu(mean_G(u2e[nodes]) @ ge_W + ge_b)
// One WG per 64 batch rows.
// ---------------------------------------------------------------------------
__global__ __launch_bounds__(256)
void group_embed_kernel(const int* __restrict__ nodes, const float* __restrict__ u2e,
                        const _Float16* __restrict__ GEt, const float* __restrict__ geb,
                        _Float16* __restrict__ Gall) {
    __shared__ _Float16 GM[64 * 128];
    int b0 = blockIdx.x * 64;
    int tid = threadIdx.x;
    for (int i = tid; i < 64 * 128; i += 256) {
        int m = i >> 7, d = i & 127;
        const int* nrow = nodes + (size_t)(b0 + m) * GMEM;
        float s = 0.0f;
#pragma unroll
        for (int g = 0; g < GMEM; ++g)
            s += u2e[(size_t)nrow[g] * DDIM + d];
        GM[i] = (_Float16)(s * 0.125f);
    }
    __syncthreads();

    int wave = tid >> 5, lane = tid & 31;
    int ncol = wave * 16 + (lane & 15);
    const _Float16* wp = GEt + (size_t)ncol * 128;
    v8f acc[4] = {};
    for (int k0 = 0; k0 < 128; k0 += 32) {
        v16h bf = load_fragB(wp, k0, lane);
        v16h af[4];
#pragma unroll
        for (int mt = 0; mt < 4; ++mt)
            af[mt] = load_fragA(GM + (mt * 16 + (lane & 15)) * 128, k0, lane);
#pragma unroll
        for (int mt = 0; mt < 4; ++mt)
            acc[mt] = __builtin_amdgcn_wmma_f32_16x16x32_f16(
                false, af[mt], false, bf, (short)0, acc[mt], false, false);
    }
    float bv = geb[ncol];
#pragma unroll
    for (int mt = 0; mt < 4; ++mt) {
        int mb = mt * 16 + ((lane & 16) >> 1);
#pragma unroll
        for (int r = 0; r < 8; ++r)
            Gall[(size_t)(b0 + mb + r) * DDIM + ncol] =
                (_Float16)fmaxf(acc[mt][r] + bv, 0.0f);
    }
}

// ---------------------------------------------------------------------------
// Kernel 2: per-batch history MLP + attention.  One 256-thread WG per batch.
// ---------------------------------------------------------------------------
__global__ __launch_bounds__(256)
void history_attend_kernel(
        const int* __restrict__ history_u, const int* __restrict__ history_r,
        const int* __restrict__ hist_len,
        const float* __restrict__ v2e, const float* __restrict__ r2e,
        const _Float16* __restrict__ W1t, const float* __restrict__ b1,
        const _Float16* __restrict__ W2t, const float* __restrict__ b2,
        const _Float16* __restrict__ A1t, const float* __restrict__ a1b,
        const _Float16* __restrict__ A2t, const float* __restrict__ a2b,
        const float* __restrict__ w3, const float* __restrict__ w3b,
        const _Float16* __restrict__ Gall,
        float* __restrict__ out) {
    __shared__ _Float16 X2[64 * 256];   // X = [e_uv | e_r]; later [O | g_rep]
    __shared__ _Float16 Hs[64 * 128];   // H; later A1
    __shared__ _Float16 gbuf[128];
    __shared__ float logits[64];
    __shared__ float att[64];

    int b = blockIdx.x;
    int tid = threadIdx.x;
    int wave = tid >> 5, lane = tid & 31;

    if (tid < 64)  logits[tid] = w3b[0];                   // init with att3 bias
    if (tid < 128) gbuf[tid] = Gall[(size_t)b * DDIM + tid];

    // Gather e_uv / e_r -> f16 concat in LDS
    const int* hu = history_u + (size_t)b * LHIST;
    const int* hr = history_r + (size_t)b * LHIST;
    for (int i = tid * 4; i < 64 * 256; i += 256 * 4) {
        int m = i >> 8, c = i & 255;
        const float* src = (c < 128) ? (v2e + (size_t)hu[m] * DDIM + c)
                                     : (r2e + (size_t)hr[m] * DDIM + (c - 128));
        float4 f = *(const float4*)src;
        _Float16* dst = X2 + i;
        dst[0] = (_Float16)f.x; dst[1] = (_Float16)f.y;
        dst[2] = (_Float16)f.z; dst[3] = (_Float16)f.w;
    }
    __syncthreads();

    // H = relu(X @ W1 + b1)                  [64,256]x[256,128]
    gemm_relu_store(X2, 256, 256, W1t, b1, Hs, 128, wave, lane);
    __syncthreads();

    // Broadcast g into concat cols 128..255 (X region is dead now)
    for (int i = tid; i < 64 * 128; i += 256) {
        int m = i >> 7, d = i & 127;
        X2[m * 256 + 128 + d] = gbuf[d];
    }
    // O = relu(H @ W2 + b2) -> concat cols 0..127
    gemm_relu_store(Hs, 128, 128, W2t, b2, X2, 256, wave, lane);
    __syncthreads();

    // A1 = relu([O|g] @ att1 + a1b)          [64,256]x[256,128]
    gemm_relu_store(X2, 256, 256, A1t, a1b, Hs, 128, wave, lane);
    __syncthreads();

    // A2 = relu(A1 @ att2 + a2b), fused with logits = A2 @ att3_W
    {
        int ncol = wave * 16 + (lane & 15);
        const _Float16* wp = A2t + (size_t)ncol * 128;
        v8f acc[4] = {};
        for (int k0 = 0; k0 < 128; k0 += 32) {
            v16h bf = load_fragB(wp, k0, lane);
            v16h af[4];
#pragma unroll
            for (int mt = 0; mt < 4; ++mt)
                af[mt] = load_fragA(Hs + (mt * 16 + (lane & 15)) * 128, k0, lane);
#pragma unroll
            for (int mt = 0; mt < 4; ++mt)
                acc[mt] = __builtin_amdgcn_wmma_f32_16x16x32_f16(
                    false, af[mt], false, bf, (short)0, acc[mt], false, false);
        }
        float bv = a2b[ncol];
        float wv = w3[ncol];
#pragma unroll
        for (int mt = 0; mt < 4; ++mt) {
            int mb = mt * 16 + ((lane & 16) >> 1);
#pragma unroll
            for (int r = 0; r < 8; ++r) {
                float v = fmaxf(acc[mt][r] + bv, 0.0f) * wv;
                v += __shfl_xor(v, 1);
                v += __shfl_xor(v, 2);
                v += __shfl_xor(v, 4);
                v += __shfl_xor(v, 8);           // sum over the 16 N-lanes
                if ((lane & 15) == 0) atomicAdd(&logits[mb + r], v);
            }
        }
    }
    __syncthreads();

    // Masked softmax over L=64 (lengths = hist_len+1 >= 1), by wave 0
    if (wave == 0) {
        int len = hist_len[b] + 1;
        float l0 = (lane      < len) ? logits[lane]      : -1e30f;
        float l1 = (lane + 32 < len) ? logits[lane + 32] : -1e30f;
        float mx = fmaxf(l0, l1);
        for (int off = 1; off < 32; off <<= 1) mx = fmaxf(mx, __shfl_xor(mx, off));
        float e0 = (lane      < len) ? __expf(l0 - mx) : 0.0f;
        float e1 = (lane + 32 < len) ? __expf(l1 - mx) : 0.0f;
        float s = e0 + e1;
        for (int off = 1; off < 32; off <<= 1) s += __shfl_xor(s, off);
        float inv = 1.0f / s;
        att[lane]      = e0 * inv;
        att[lane + 32] = e1 * inv;
    }
    __syncthreads();

    // out[b,d] = sum_l att[l] * O[l,d]   (O is f16 in concat cols 0..127)
    if (tid < 128) {
        float s = 0.0f;
#pragma unroll 8
        for (int l = 0; l < 64; ++l)
            s += att[l] * (float)X2[l * 256 + tid];
        out[(size_t)b * DDIM + tid] = s;
    }
}

// ---------------------------------------------------------------------------
extern "C" void kernel_launch(void* const* d_in, const int* in_sizes, int n_in,
                              void* d_out, int out_size, void* d_ws, size_t ws_size,
                              hipStream_t stream) {
    (void)in_sizes; (void)n_in; (void)out_size; (void)ws_size;
    const int*   nodes     = (const int*)d_in[0];
    const int*   history_u = (const int*)d_in[1];
    const int*   history_r = (const int*)d_in[2];
    const int*   hist_len  = (const int*)d_in[3];
    const float* v2e = (const float*)d_in[4];
    const float* u2e = (const float*)d_in[5];
    const float* r2e = (const float*)d_in[6];
    const float* W1  = (const float*)d_in[7];
    const float* b1  = (const float*)d_in[8];
    const float* W2  = (const float*)d_in[9];
    const float* b2  = (const float*)d_in[10];
    const float* a1W = (const float*)d_in[11];
    const float* a1b = (const float*)d_in[12];
    const float* a2W = (const float*)d_in[13];
    const float* a2b = (const float*)d_in[14];
    const float* a3W = (const float*)d_in[15];
    const float* a3b = (const float*)d_in[16];
    const float* geW = (const float*)d_in[17];
    const float* geb = (const float*)d_in[18];

    _Float16* ws   = (_Float16*)d_ws;
    _Float16* W1t  = ws;
    _Float16* W2t  = ws + 32768;
    _Float16* A1t  = ws + 49152;
    _Float16* A2t  = ws + 81920;
    _Float16* GEt  = ws + 98304;
    _Float16* Gall = ws + 114688;

    transpose_f32_to_f16<<<(256 * 128 + 255) / 256, 256, 0, stream>>>(W1,  W1t, 256, 128);
    transpose_f32_to_f16<<<(128 * 128 + 255) / 256, 256, 0, stream>>>(W2,  W2t, 128, 128);
    transpose_f32_to_f16<<<(256 * 128 + 255) / 256, 256, 0, stream>>>(a1W, A1t, 256, 128);
    transpose_f32_to_f16<<<(128 * 128 + 255) / 256, 256, 0, stream>>>(a2W, A2t, 128, 128);
    transpose_f32_to_f16<<<(128 * 128 + 255) / 256, 256, 0, stream>>>(geW, GEt, 128, 128);

    group_embed_kernel<<<BATCH / 64, 256, 0, stream>>>(nodes, u2e, GEt, geb, Gall);

    history_attend_kernel<<<BATCH, 256, 0, stream>>>(
        history_u, history_r, hist_len, v2e, r2e,
        W1t, b1, W2t, b2, A1t, a1b, A2t, a2b, a3W, a3b, Gall,
        (float*)d_out);
}
